// SelfAttention_23304492548162
// MI455X (gfx1250) — matrix operand order repaired and checked by reading, not compiled, not run on hardware
//
#include <hip/hip_runtime.h>

// ---------------------------------------------------------------------------
// Self-attention forward for MI455X (gfx1250, wave32, WMMA bf16).
// B=4, S=2048, D=1024, H=16, HD=64.  Compute-bound (137 GFLOP vs ~150MB HBM):
// all matmuls via v_wmma_f32_16x16x32_bf16.  GEMM waves hold 32x64 register
// tiles (8 wmma / K-step).  Flash attention double-buffers K/V^T tiles in LDS
// via GLOBAL_LOAD_ASYNC_TO_LDS_B128 (ASYNCcnt) when the builtin is available.
// ---------------------------------------------------------------------------

#define BB  4
#define SS  2048
#define DD  1024
#define HH  16
#define HDD 64

typedef __attribute__((ext_vector_type(16))) __bf16 v16bf;
typedef __attribute__((ext_vector_type(8)))  float  v8f;
typedef __attribute__((ext_vector_type(4)))  int    v4i;

union Frag16 {
  v16bf v;
  uint4 q[2];
  unsigned short u[16];
};

__device__ __forceinline__ unsigned short f2bf(float f) {
  unsigned int u = __float_as_uint(f);
  u += 0x7FFFu + ((u >> 16) & 1u);   // round-to-nearest-even
  return (unsigned short)(u >> 16);
}

__device__ __forceinline__ v8f wmma_bf16(v16bf a, v16bf b, v8f c) {
  // (neg_a, A, neg_b, B, c_mod, C, reuse_a, reuse_b)
  return __builtin_amdgcn_wmma_f32_16x16x32_bf16(false, a, false, b,
                                                 (short)0, c, false, false);
}

// ---- CDNA5 async global->LDS copy (ASYNCcnt) with sync fallback -----------
#if defined(__has_builtin)
#  if __has_builtin(__builtin_amdgcn_global_load_async_to_lds_b128)
#    define HAS_ASYNC_LDS 1
#  else
#    define HAS_ASYNC_LDS 0
#  endif
#else
#  define HAS_ASYNC_LDS 0
#endif

__device__ __forceinline__ void async_copy_b128(const unsigned short* gsrc,
                                                unsigned short* ldst) {
#if HAS_ASYNC_LDS
  // Prototype (from hipcc diagnostic): (v4i as(1)*, v4i as(3)*, Imm, Imm)
  __builtin_amdgcn_global_load_async_to_lds_b128(
      (__attribute__((address_space(1))) v4i*)gsrc,
      (__attribute__((address_space(3))) v4i*)ldst, 0, 0);
#else
  *(uint4*)ldst = *(const uint4*)gsrc;
#endif
}

__device__ __forceinline__ void wait_async0() {
#if HAS_ASYNC_LDS
#  if __has_builtin(__builtin_amdgcn_s_wait_asynccnt)
  __builtin_amdgcn_s_wait_asynccnt(0);
#  else
  asm volatile("s_wait_asynccnt 0" ::: "memory");
#  endif
#endif
}

// ---------------------------------------------------------------------------
// Kernel 1: fused QKV projection.  y = x @ W + b, f32 in -> bf16 out.
// grid = (D/64, M/256, 3 [q,k,v]), block = 256 (8 waves).
// Each wave: 32M x 64N register tile = 8 accumulators, 8 wmma per K-step.
// Q,K written [B,H,S,HD]; V written transposed [B,H,HD,S].
// ---------------------------------------------------------------------------
__global__ __launch_bounds__(256) void qkv_proj_kernel(
    const float* __restrict__ x,
    const float* __restrict__ qw, const float* __restrict__ kw,
    const float* __restrict__ vw,
    const float* __restrict__ qb, const float* __restrict__ kb,
    const float* __restrict__ vb,
    unsigned short* __restrict__ Qo, unsigned short* __restrict__ Ko,
    unsigned short* __restrict__ Vo)
{
  const int mat = blockIdx.z;
  const float* w    = (mat == 0) ? qw : (mat == 1) ? kw : vw;
  const float* bias = (mat == 0) ? qb : (mat == 1) ? kb : vb;
  unsigned short* outp = (mat == 0) ? Qo : (mat == 1) ? Ko : Vo;

  const int n0 = blockIdx.x * 64;
  const int m0 = blockIdx.y * 256;
  const int tid  = threadIdx.x;
  const int lane = tid & 31;
  const int wave = tid >> 5;
  const int lrow = lane & 15;
  const int lhi  = lane >> 4;         // 0 or 1

  __shared__ __align__(16) unsigned short xs[256 * 32];  // [m][k] bf16
  __shared__ __align__(16) unsigned short wst[64 * 32];  // [n][k] bf16 (transposed)

  v8f acc[2][4] = {};

  for (int k0 = 0; k0 < DD; k0 += 32) {
    __syncthreads();
    { // stage x tile (256x32): one full row per thread, f32->bf16
      const float* src = x + (size_t)(m0 + tid) * DD + k0;
      __align__(16) unsigned short tmp[32];
      #pragma unroll
      for (int i = 0; i < 32; ++i) tmp[i] = f2bf(src[i]);
      #pragma unroll
      for (int i = 0; i < 4; ++i)
        *(uint4*)&xs[tid * 32 + i * 8] = *(const uint4*)&tmp[i * 8];
    }
    { // stage w tile transposed: wst[n][k]
      int kk = tid >> 3;
      int nn = (tid & 7) * 8;
      const float* src = w + (size_t)(k0 + kk) * DD + (n0 + nn);
      #pragma unroll
      for (int i = 0; i < 8; ++i) wst[(nn + i) * 32 + kk] = f2bf(src[i]);
    }
    __syncthreads();

    Frag16 a[2];  // A: 16x32 bf16, lane=row, two 8-K chunks
    #pragma unroll
    for (int ms = 0; ms < 2; ++ms) {
      int row = wave * 32 + ms * 16 + lrow;
      int kb0 = lhi * 16;
      a[ms].q[0] = *(const uint4*)&xs[row * 32 + kb0];
      a[ms].q[1] = *(const uint4*)&xs[row * 32 + kb0 + 8];
    }
    #pragma unroll
    for (int ns = 0; ns < 4; ++ns) {  // B: 32x16, lane=col, 16 contiguous K
      Frag16 bfr;
      int nrow = ns * 16 + lrow;
      int kb2  = lhi * 16;
      bfr.q[0] = *(const uint4*)&wst[nrow * 32 + kb2];
      bfr.q[1] = *(const uint4*)&wst[nrow * 32 + kb2 + 8];
      #pragma unroll
      for (int ms = 0; ms < 2; ++ms)
        acc[ms][ns] = wmma_bf16(a[ms].v, bfr.v, acc[ms][ns]);
    }
  }

  #pragma unroll
  for (int ms = 0; ms < 2; ++ms) {
    #pragma unroll
    for (int ns = 0; ns < 4; ++ns) {
      int n = n0 + ns * 16 + lrow;
      float bv = bias[n];
      int h = n >> 6, hd = n & 63;
      #pragma unroll
      for (int r = 0; r < 8; ++r) {
        int m  = m0 + wave * 32 + ms * 16 + lhi * 8 + r;
        int bb = m >> 11;            // / S
        int s  = m & (SS - 1);
        unsigned short val = f2bf(acc[ms][ns][r] + bv);
        if (mat == 2)  // V transposed [B,H,HD,S]
          outp[(((size_t)bb * HH + h) * HDD + hd) * SS + s] = val;
        else           // Q,K [B,H,S,HD]
          outp[(((size_t)bb * HH + h) * SS + s) * HDD + hd] = val;
      }
    }
  }
}

// ---------------------------------------------------------------------------
// Kernel 2: flash attention.  grid = (S/64, H, B), block = 128 (4 waves).
// Wave owns 16 q rows.  Scores computed transposed: S^T = K @ Q^T, so each
// lane owns one q row -> softmax row reduce = 8 reg ops + one shfl_xor(16).
// K / V^T tiles (32 kv) double-buffered in LDS via async global->LDS loads.
// ---------------------------------------------------------------------------
__global__ __launch_bounds__(128) void flash_attn_kernel(
    const unsigned short* __restrict__ Qb,
    const unsigned short* __restrict__ Kb,
    const unsigned short* __restrict__ Vtb,
    unsigned short* __restrict__ Attb)
{
  const int qt = blockIdx.x;
  const int h  = blockIdx.y;
  const int b  = blockIdx.z;
  const int tid  = threadIdx.x;
  const int lane = tid & 31;
  const int wave = tid >> 5;
  const int lrow = lane & 15;
  const int lhi  = lane >> 4;

  const size_t bh = (size_t)b * HH + h;
  const int q0 = qt * 64 + wave * 16;

  __shared__ __align__(16) unsigned short Ksm[2][32 * 64];  // [kv][hd]
  __shared__ __align__(16) unsigned short Vsm[2][64 * 32];  // V^T: [hd][kv]

  // Per-thread staging coordinates (16 elements each, two b128 copies).
  const int rK = (tid * 16) >> 6, cK = (tid * 16) & 63;
  const int rV = (tid * 16) >> 5, cV = (tid * 16) & 31;

  // Q^T B-fragments (K-dim = hd), resident whole kernel.
  Frag16 qf0, qf1;
  {
    const unsigned short* qrow = Qb + (bh * SS + (q0 + lrow)) * HDD + lhi * 16;
    qf0.q[0] = *(const uint4*)(qrow + 0);
    qf0.q[1] = *(const uint4*)(qrow + 8);
    qf1.q[0] = *(const uint4*)(qrow + 32);
    qf1.q[1] = *(const uint4*)(qrow + 40);
  }

  v8f acc0 = {}, acc1 = {}, acc2 = {}, acc3 = {};   // out^T, hd tiles 0..3
  float m_run = -__builtin_inff();
  float l_run = 0.0f;
  const float Cs = 0.125f * 1.44269504088896340736f; // 1/sqrt(HD) * log2(e)

  auto issue_tile = [&](int kv0, int buf) {
    const unsigned short* ksrc = Kb + (bh * SS + kv0 + rK) * HDD + cK;
    async_copy_b128(ksrc,     &Ksm[buf][rK * 64 + cK]);
    async_copy_b128(ksrc + 8, &Ksm[buf][rK * 64 + cK + 8]);
    const unsigned short* vsrc = Vtb + (bh * HDD + rV) * SS + kv0 + cV;
    async_copy_b128(vsrc,     &Vsm[buf][rV * 32 + cV]);
    async_copy_b128(vsrc + 8, &Vsm[buf][rV * 32 + cV + 8]);
  };

  issue_tile(0, 0);  // prologue prefetch

  for (int kv0 = 0; kv0 < SS; kv0 += 32) {
    const int p = (kv0 >> 5) & 1;
    wait_async0();        // tile kv0 has landed (this wave's copies)
    __syncthreads();      // ... and everyone else's; prev buffer free
    if (kv0 + 32 < SS) issue_tile(kv0 + 32, p ^ 1);   // overlap with compute

    // S^T tiles: st0 -> kv 0..15, st1 -> kv 16..31 (M=kv, N=q)
    v8f st0 = {}, st1 = {};
    #pragma unroll
    for (int step = 0; step < 2; ++step) {
      Frag16 kf;
      int hb = step * 32 + lhi * 16;
      kf.q[0] = *(const uint4*)&Ksm[p][lrow * 64 + hb];
      kf.q[1] = *(const uint4*)&Ksm[p][lrow * 64 + hb + 8];
      st0 = wmma_bf16(kf.v, (step ? qf1.v : qf0.v), st0);
      kf.q[0] = *(const uint4*)&Ksm[p][(16 + lrow) * 64 + hb];
      kf.q[1] = *(const uint4*)&Ksm[p][(16 + lrow) * 64 + hb + 8];
      st1 = wmma_bf16(kf.v, (step ? qf1.v : qf0.v), st1);
    }

    // ---- online softmax (per-lane q row) ----
    float pv[16];
    float vmax = -__builtin_inff();
    #pragma unroll
    for (int i = 0; i < 8; ++i) {
      float t0 = st0[i] * Cs;
      float t1 = st1[i] * Cs;
      pv[i] = t0; pv[8 + i] = t1;
      vmax = fmaxf(vmax, fmaxf(t0, t1));
    }
    vmax = fmaxf(vmax, __shfl_xor(vmax, 16, 32));
    float m_new = fmaxf(m_run, vmax);
    float alpha = exp2f(m_run - m_new);
    float rsum = 0.0f;
    #pragma unroll
    for (int i = 0; i < 16; ++i) { pv[i] = exp2f(pv[i] - m_new); rsum += pv[i]; }
    rsum += __shfl_xor(rsum, 16, 32);
    l_run = l_run * alpha + rsum;
    m_run = m_new;
    #pragma unroll
    for (int r = 0; r < 8; ++r) {
      acc0[r] *= alpha; acc1[r] *= alpha; acc2[r] *= alpha; acc3[r] *= alpha;
    }

    // Repack P^T (C-layout) into B-layout (32kv x 16q): one xor-16 exchange.
    Frag16 pf;
    #pragma unroll
    for (int j = 0; j < 8; ++j) {
      float o0 = __shfl_xor(pv[j], 16, 32);
      float o1 = __shfl_xor(pv[8 + j], 16, 32);
      float e0 = lhi ? o1 : pv[j];       // element j   : kv = (lhi?16:0)+j
      float e1 = lhi ? pv[8 + j] : o0;   // element j+8 : kv = (lhi?16:0)+8+j
      pf.u[j]     = f2bf(e0);
      pf.u[8 + j] = f2bf(e1);
    }

    // out^T += V^T @ P^T  (4 hd tiles of 16)
    #pragma unroll
    for (int f = 0; f < 4; ++f) {
      Frag16 vf;
      int hr = f * 16 + lrow;
      int cb = lhi * 16;
      vf.q[0] = *(const uint4*)&Vsm[p][hr * 32 + cb];
      vf.q[1] = *(const uint4*)&Vsm[p][hr * 32 + cb + 8];
      v8f& a = (f == 0) ? acc0 : (f == 1) ? acc1 : (f == 2) ? acc2 : acc3;
      a = wmma_bf16(vf.v, pf.v, a);
    }
  }

  // epilogue: lane q row, regs = hd; write [B,S,D] with d = h*64+hd
  float inv_l = 1.0f / l_run;
  unsigned short* dst = Attb + ((size_t)b * SS + (q0 + lrow)) * DD + h * HDD;
  #pragma unroll
  for (int f = 0; f < 4; ++f) {
    v8f a = (f == 0) ? acc0 : (f == 1) ? acc1 : (f == 2) ? acc2 : acc3;
    #pragma unroll
    for (int r = 0; r < 8; ++r)
      dst[f * 16 + lhi * 8 + r] = f2bf(a[r] * inv_l);
  }
}

// ---------------------------------------------------------------------------
// Kernel 3: output projection.  out = att(bf16) @ o_w + o_b, f32 out.
// grid = (D/64, M/256), block = 256.  Same 32x64 per-wave register tiling.
// ---------------------------------------------------------------------------
__global__ __launch_bounds__(256) void o_proj_kernel(
    const unsigned short* __restrict__ att,
    const float* __restrict__ ow, const float* __restrict__ ob,
    float* __restrict__ out)
{
  const int n0 = blockIdx.x * 64;
  const int m0 = blockIdx.y * 256;
  const int tid  = threadIdx.x;
  const int lane = tid & 31;
  const int wave = tid >> 5;
  const int lrow = lane & 15;
  const int lhi  = lane >> 4;

  __shared__ __align__(16) unsigned short as_[256 * 32];
  __shared__ __align__(16) unsigned short wst[64 * 32];

  v8f acc[2][4] = {};

  for (int k0 = 0; k0 < DD; k0 += 32) {
    __syncthreads();
    { // att tile already bf16: one row per thread, 4x b128 copies
      const unsigned short* src = att + (size_t)(m0 + tid) * DD + k0;
      #pragma unroll
      for (int i = 0; i < 4; ++i)
        *(uint4*)&as_[tid * 32 + i * 8] = *(const uint4*)(src + i * 8);
    }
    { // weight tile transposed, f32->bf16
      int kk = tid >> 3;
      int nn = (tid & 7) * 8;
      const float* src = ow + (size_t)(k0 + kk) * DD + (n0 + nn);
      #pragma unroll
      for (int i = 0; i < 8; ++i) wst[(nn + i) * 32 + kk] = f2bf(src[i]);
    }
    __syncthreads();

    Frag16 a[2];
    #pragma unroll
    for (int ms = 0; ms < 2; ++ms) {
      int row = wave * 32 + ms * 16 + lrow;
      int kb0 = lhi * 16;
      a[ms].q[0] = *(const uint4*)&as_[row * 32 + kb0];
      a[ms].q[1] = *(const uint4*)&as_[row * 32 + kb0 + 8];
    }
    #pragma unroll
    for (int ns = 0; ns < 4; ++ns) {
      Frag16 bfr;
      int nrow = ns * 16 + lrow;
      int kb2  = lhi * 16;
      bfr.q[0] = *(const uint4*)&wst[nrow * 32 + kb2];
      bfr.q[1] = *(const uint4*)&wst[nrow * 32 + kb2 + 8];
      #pragma unroll
      for (int ms = 0; ms < 2; ++ms)
        acc[ms][ns] = wmma_bf16(a[ms].v, bfr.v, acc[ms][ns]);
    }
  }

  #pragma unroll
  for (int ms = 0; ms < 2; ++ms) {
    #pragma unroll
    for (int ns = 0; ns < 4; ++ns) {
      int n = n0 + ns * 16 + lrow;
      float bv = ob[n];
      #pragma unroll
      for (int r = 0; r < 8; ++r) {
        int m = m0 + wave * 32 + ms * 16 + lhi * 8 + r;
        out[(size_t)m * DD + n] = acc[ms][ns][r] + bv;
      }
    }
  }
}

// ---------------------------------------------------------------------------
extern "C" void kernel_launch(void* const* d_in, const int* in_sizes, int n_in,
                              void* d_out, int out_size, void* d_ws,
                              size_t ws_size, hipStream_t stream) {
  (void)in_sizes; (void)n_in; (void)out_size; (void)ws_size;
  const float* x   = (const float*)d_in[0];
  const float* q_w = (const float*)d_in[1];
  const float* q_b = (const float*)d_in[2];
  const float* k_w = (const float*)d_in[3];
  const float* k_b = (const float*)d_in[4];
  const float* v_w = (const float*)d_in[5];
  const float* v_b = (const float*)d_in[6];
  const float* o_w = (const float*)d_in[7];
  const float* o_b = (const float*)d_in[8];
  float* out = (float*)d_out;

  const size_t BSD = (size_t)BB * SS * DD;     // 8,388,608 elements
  unsigned short* Qb   = (unsigned short*)d_ws;         // [B,H,S,HD] bf16
  unsigned short* Kb   = Qb + BSD;                      // [B,H,S,HD] bf16
  unsigned short* Vtb  = Qb + 2 * BSD;                  // [B,H,HD,S] bf16
  unsigned short* Attb = Qb + 3 * BSD;                  // [B,S,D]    bf16

  qkv_proj_kernel<<<dim3(DD / 64, (BB * SS) / 256, 3), 256, 0, stream>>>(
      x, q_w, k_w, v_w, q_b, k_b, v_b, Qb, Kb, Vtb);

  flash_attn_kernel<<<dim3(SS / 64, HH, BB), 128, 0, stream>>>(
      Qb, Kb, Vtb, Attb);

  o_proj_kernel<<<dim3(DD / 64, (BB * SS) / 256), 256, 0, stream>>>(
      Attb, o_w, o_b, out);
}
